// VectorQuantizerClass_77695958385279
// MI455X (gfx1250) — matrix operand order repaired and checked by reading, not compiled, not run on hardware
//
#include <hip/hip_runtime.h>
#include <hip/hip_bf16.h>
#include <float.h>
#include <math.h>

// Problem constants (from reference)
#define NPTS 131072
#define KCB  1024
#define DIM  256

typedef __attribute__((ext_vector_type(16))) __bf16 v16bf;
typedef __attribute__((ext_vector_type(8)))  float  v8f;
typedef __attribute__((ext_vector_type(4)))  unsigned int u32x4;
typedef __attribute__((ext_vector_type(4)))  int i32x4;
typedef __attribute__((ext_vector_type(8)))  int i32x8;

// ---- workspace layout (bytes) ----
#define WB_OFF     0u                          // KCB*DIM bf16 = 524288 B
#define CNORM_OFF  (KCB * DIM * 2u)            // 524288: KCB floats
#define COUNTS_OFF (CNORM_OFF + KCB * 4u)      // 528384: KCB floats
#define PARTS_OFF  (COUNTS_OFF + KCB * 4u)     // 532480: NPTS*2 floats (per-block loss partials)
// total ~1.58 MB

// ---- output layout (floats) ----
// [0] loss | [1 .. N*D] quantized_st | [1+N*D] perplexity | [2+N*D .. +N] close_idx
#define OUT_LOSS 0
#define OUT_Q    ((size_t)1)
#define OUT_P    ((size_t)1 + (size_t)NPTS * DIM)
#define OUT_CI   ((size_t)2 + (size_t)NPTS * DIM)

// ---------------- helpers ----------------
__device__ inline float2 blockReduceSum2(float a, float b) {
  __shared__ float sa[32], sb[32];
  const int lane = threadIdx.x & 31, w = threadIdx.x >> 5;
#pragma unroll
  for (int o = 16; o > 0; o >>= 1) {
    a += __shfl_down(a, o, 32);
    b += __shfl_down(b, o, 32);
  }
  if (lane == 0) { sa[w] = a; sb[w] = b; }
  __syncthreads();
  const int nw = (blockDim.x + 31) >> 5;
  a = (threadIdx.x < nw) ? sa[threadIdx.x] : 0.0f;
  b = (threadIdx.x < nw) ? sb[threadIdx.x] : 0.0f;
  if (w == 0) {
#pragma unroll
    for (int o = 16; o > 0; o >>= 1) {
      a += __shfl_down(a, o, 32);
      b += __shfl_down(b, o, 32);
    }
  }
  return make_float2(a, b);  // valid in thread 0
}

// ---------------- kernels ----------------

// zero the histogram bins
__global__ __launch_bounds__(1024) void vq_init(float* counts) {
  const int t = threadIdx.x;
  if (t < KCB) counts[t] = 0.0f;
}

// W (fp32) -> Wb (bf16) + cnorm[k] = ||W_k||^2 (fp32)
__global__ __launch_bounds__(256) void vq_prep(const float* __restrict__ W,
                                               __bf16* __restrict__ Wb,
                                               float* __restrict__ cnorm) {
  const int k = blockIdx.x, t = threadIdx.x;
  const float w = W[(size_t)k * DIM + t];
  Wb[(size_t)k * DIM + t] = (__bf16)w;
  float2 s = blockReduceSum2(w * w, 0.0f);
  if (t == 0) cnorm[k] = s.x;
}

// histogram of y (float adds of integers: exact, order-independent)
__global__ __launch_bounds__(256) void vq_hist(const int* __restrict__ y,
                                               float* __restrict__ counts) {
  const int i = blockIdx.x * blockDim.x + threadIdx.x;
  atomicAdd(&counts[y[i]], 1.0f);
}

// Nearest-code search: per wave, one 16-row tile of x vs all 1024 codes.
// B tiles (16 codes x 256 dims bf16 = 8 KB, contiguous in Wb) are staged into
// LDS once per block via the Tensor Data Mover, double-buffered, so the 8 waves
// share one DMA stream and the WMMAs read via ds_load_b128.
// dist ranking uses  ||c||^2 - 2 x.c   (per-row ||x||^2 dropped — constant under argmin)
__global__ __launch_bounds__(256) void vq_argmin(const float* __restrict__ x,
                                                 const __bf16* __restrict__ Wb,
                                                 const float* __restrict__ cnorm,
                                                 float* __restrict__ out) {
  __shared__ __align__(128) __bf16 btile[2][16 * DIM];  // 2 x 8 KB double buffer

  const int lane    = threadIdx.x & 31;
  const int wv      = threadIdx.x >> 5;
  const int rowTile = blockIdx.x * 8 + wv;   // 8 waves per block
  const int m0      = rowTile * 16;

  const int halfSel = lane >> 4;   // 0: lanes 0-15, 1: lanes 16-31
  const int l16     = lane & 15;

  // Issue one TDM load: Wb tile `nt` (16 rows x 256 bf16) -> btile[buf].
  // D# built per CDNA5 ISA ch. 8.3/8.4 (2D tensor, data_size=2B, no pad/iterate,
  // workgroup_mask=0: not in a cluster).
  auto issue_tdm = [&](int nt, int buf) {
    const unsigned lds_off = (unsigned)(size_t)(void*)&btile[buf][0];
    const unsigned long long ga =
        (unsigned long long)(const void*)(Wb + (size_t)nt * 16 * DIM);
    u32x4 g0 = { 1u,                                   // count=1, user descriptor
                 lds_off,                              // lds_addr [63:32]
                 (unsigned)(ga & 0xFFFFFFFFull),       // global_addr [95:64]
                 ((unsigned)((ga >> 32) & 0x01FFFFFFull)) | 0x80000000u };  // addr[56:32] | type=2
    i32x8 g1 = { (int)0x00010000u,                     // data_size=1 (2 bytes/elem)
                 (int)(((unsigned)DIM & 0xFFFFu) << 16),      // tensor_dim0[15:0] = 256
                 (int)(16u << 16),                            // tensor_dim1[15:0] = 16 rows
                 (int)((unsigned)DIM << 16),                  // tile_dim0 = 256
                 (int)16,                                      // tile_dim1 = 16 (tile_dim2=0)
                 (int)DIM,                                     // tensor_dim0_stride = 256
                 0, 0 };
    i32x4 z4 = { 0, 0, 0, 0 };
#if __has_include(<hip/amd_detail/amd_gfx1250_TDM.h>)
    i32x8 z8 = { 0, 0, 0, 0, 0, 0, 0, 0 };
    __builtin_amdgcn_tensor_load_to_lds(g0, g1, z4, z4, z8, 0);  // 6-arg toolchain
#else
    __builtin_amdgcn_tensor_load_to_lds(g0, g1, z4, z4, 0);      // 5-arg toolchain
#endif
  };

  // --- A fragments: rows m0..m0+15, full D=256, bf16, held in registers ---
  // A k order used here (contiguous, same as B): lanes 0-15 hold K 0..15; lanes 16-31 hold K 16..31 (matches B)
  v16bf afrag[8];
  {
    const float* xr0 = x + (size_t)(m0 + l16) * DIM;
#pragma unroll
    for (int kt = 0; kt < 8; ++kt) {
      const float* p = xr0 + kt * 32 + halfSel * 16;
      float4 c0 = ((const float4*)p)[0];
      float4 c1 = ((const float4*)p)[1];
      float4 c2 = ((const float4*)(p + 8))[0];
      float4 c3 = ((const float4*)(p + 8))[1];
      v16bf a;
      a[0]  = (__bf16)c0.x; a[1]  = (__bf16)c0.y; a[2]  = (__bf16)c0.z; a[3]  = (__bf16)c0.w;
      a[4]  = (__bf16)c1.x; a[5]  = (__bf16)c1.y; a[6]  = (__bf16)c1.z; a[7]  = (__bf16)c1.w;
      a[8]  = (__bf16)c2.x; a[9]  = (__bf16)c2.y; a[10] = (__bf16)c2.z; a[11] = (__bf16)c2.w;
      a[12] = (__bf16)c3.x; a[13] = (__bf16)c3.y; a[14] = (__bf16)c3.z; a[15] = (__bf16)c3.w;
      afrag[kt] = a;
    }
  }

  float minv[8];
  int   mini[8];
#pragma unroll
  for (int v = 0; v < 8; ++v) { minv[v] = FLT_MAX; mini[v] = 0; }

  const int bcol = l16;            // code within the 16-wide N tile
  const int bko  = halfSel * 16;   // B layout: lanes 0-15 carry K 0..15, lanes 16-31 K 16..31

  // Prologue: DMA tile 0 into buffer 0 (wave 0 only; TDM ignores EXEC)
  if (wv == 0) issue_tdm(0, 0);

#pragma unroll 1
  for (int nt = 0; nt < KCB / 16; ++nt) {
    if (wv == 0) __builtin_amdgcn_s_wait_tensorcnt(0);  // tile nt landed in LDS
    __syncthreads();                                     // publish to all 8 waves
    if (wv == 0 && nt + 1 < KCB / 16)                    // overlap next DMA with compute
      issue_tdm(nt + 1, (nt + 1) & 1);

    const __bf16* wrow = &btile[nt & 1][(size_t)bcol * DIM + bko];
    v16bf bfr[8];
#pragma unroll
    for (int kt = 0; kt < 8; ++kt)                       // 2x ds_load_b128 each
      bfr[kt] = *(const v16bf*)(wrow + kt * 32);

    v8f acc = {};
#pragma unroll
    for (int kt = 0; kt < 8; ++kt)
      acc = __builtin_amdgcn_wmma_f32_16x16x32_bf16(
          /*neg_a=*/false, afrag[kt], /*neg_b=*/false, bfr[kt],
          /*c_mod=*/(short)0, acc, /*reuse_a=*/false, /*reuse_b=*/false);

    const int   code = nt * 16 + bcol;
    const float cn   = cnorm[code];
#pragma unroll
    for (int v = 0; v < 8; ++v) {
      const float d = cn - 2.0f * acc[v];
      if (d < minv[v]) { minv[v] = d; mini[v] = code; }
    }
  }

  // C/D layout: VGPR v -> row (v + 8*halfSel); reduce argmin across the 16 lanes of each half
#pragma unroll
  for (int v = 0; v < 8; ++v) {
    float mv = minv[v];
    int   mi = mini[v];
#pragma unroll
    for (int off = 1; off < 16; off <<= 1) {
      const float ov = __shfl_xor(mv, off, 32);
      const int   oi = __shfl_xor(mi, off, 32);
      if (ov < mv || (ov == mv && oi < mi)) { mv = ov; mi = oi; }  // first-occurrence tie-break
    }
    minv[v] = mv; mini[v] = mi;
  }

  if (l16 == 0) {  // lane 0 holds rows 0..7, lane 16 holds rows 8..15
#pragma unroll
    for (int v = 0; v < 8; ++v) {
      const int row = m0 + v + halfSel * 8;
      out[OUT_CI + row] = (float)mini[v];
    }
  }
}

// quantized output + per-block loss partials (deterministic two-pass reduction)
__global__ __launch_bounds__(256) void vq_loss(const float* __restrict__ x,
                                               const int* __restrict__ y,
                                               const float* __restrict__ W,
                                               float* __restrict__ out,
                                               float* __restrict__ parts) {
  const int i  = blockIdx.x;
  const int d  = threadIdx.x;
  const int yi = y[i];
  const int ci = (int)out[OUT_CI + i];
  const float xv = x[(size_t)i * DIM + d];
  const float wy = W[(size_t)yi * DIM + d];
  const float wc = W[(size_t)ci * DIM + d];
  out[OUT_Q + (size_t)i * DIM + d] = wy;   // quantized_st forward value == W[y]
  const float dq = wy - xv;
  float q  = dq * dq;
  float xl = 0.0f;
  if (yi != ci) { const float dx = wc - xv; xl = dx * dx; }
  float2 s = blockReduceSum2(q, xl);
  if (threadIdx.x == 0) { parts[2 * i] = s.x; parts[2 * i + 1] = s.y; }
}

// final scalars: loss (double tree-sum of partials) + perplexity
__global__ __launch_bounds__(1024) void vq_final(const float* __restrict__ counts,
                                                 const float* __restrict__ parts,
                                                 float* __restrict__ out) {
  const int t = threadIdx.x;
  double q = 0.0, xl = 0.0;
  for (int i = t; i < NPTS; i += 1024) {
    q  += (double)parts[2 * i];
    xl += (double)parts[2 * i + 1];
  }
  __shared__ double sq[32], sx[32];
  const int lane = t & 31, w = t >> 5;
#pragma unroll
  for (int o = 16; o > 0; o >>= 1) {
    q  += __shfl_down(q, o, 32);
    xl += __shfl_down(xl, o, 32);
  }
  if (lane == 0) { sq[w] = q; sx[w] = xl; }
  __syncthreads();
  if (w == 0) {
    q = sq[lane]; xl = sx[lane];
#pragma unroll
    for (int o = 16; o > 0; o >>= 1) {
      q  += __shfl_down(q, o, 32);
      xl += __shfl_down(xl, o, 32);
    }
  }
  const float p    = counts[t] * (1.0f / NPTS);
  const float term = p * logf(p + 1e-10f);
  float2 s = blockReduceSum2(term, 0.0f);
  if (t == 0) {
    out[OUT_P]    = expf(-s.x);
    out[OUT_LOSS] = (float)((1.25 * q - 1.1 * xl) / ((double)NPTS * (double)DIM));
  }
}

// ---------------- launch ----------------
extern "C" void kernel_launch(void* const* d_in, const int* in_sizes, int n_in,
                              void* d_out, int out_size, void* d_ws, size_t ws_size,
                              hipStream_t stream) {
  const float* x = (const float*)d_in[0];
  const int*   y = (const int*)d_in[1];
  const float* W = (const float*)d_in[2];
  float* out = (float*)d_out;

  char*   ws     = (char*)d_ws;
  __bf16* Wb     = (__bf16*)(ws + WB_OFF);
  float*  cnorm  = (float*)(ws + CNORM_OFF);
  float*  counts = (float*)(ws + COUNTS_OFF);
  float*  parts  = (float*)(ws + PARTS_OFF);

  vq_init  <<<1, 1024, 0, stream>>>(counts);
  vq_prep  <<<KCB, DIM, 0, stream>>>(W, Wb, cnorm);
  vq_hist  <<<NPTS / 256, 256, 0, stream>>>(y, counts);
  vq_argmin<<<NPTS / 128, 256, 0, stream>>>(x, Wb, cnorm, out);
  vq_loss  <<<NPTS, DIM, 0, stream>>>(x, y, W, out, parts);
  vq_final <<<1, 1024, 0, stream>>>(counts, parts, out);
}
